// HybridEstimatorQNN_65481071396348
// MI455X (gfx1250) — matrix operand order repaired and checked by reading, not compiled, not run on hardware
//
#include <hip/hip_runtime.h>
#include <hip/hip_bf16.h>
#include <cmath>

typedef __attribute__((ext_vector_type(16))) _Float16 v16h;
typedef __attribute__((ext_vector_type(8)))  float    v8f;

// ---------------------------------------------------------------------------
// Phase 1: pre[t*B+b, 0:16] = x[t,b,0:64] @ Wx_packed[64,16]
// Wx_packed column 4g+h = W_g[0:64, h]  (input-weight part of each gate)
// One wave handles 16-row tiles; B fragments (weights) stay in VGPRs.
// ---------------------------------------------------------------------------
__global__ __launch_bounds__(256)
void qlstm_pre_gemm(const float* __restrict__ x,
                    const float* __restrict__ Wf,
                    const float* __restrict__ Wi,
                    const float* __restrict__ Wu,
                    const float* __restrict__ Wo,
                    float* __restrict__ pre,
                    int Mtiles)
{
    const int lane   = threadIdx.x & 31;
    const int wave   = (blockIdx.x * blockDim.x + threadIdx.x) >> 5;
    const int nwaves = (gridDim.x * blockDim.x) >> 5;

    // ---- B fragments: 32x16 f16, resident across the tile loop.
    // lanes 0-15: K=0..15 of column `lane`; lanes 16-31: K=16..31 of column lane-16.
    const int ncol = lane & 15;                 // output column 0..15 -> gate = ncol/4
    const float* Wp = (ncol < 4) ? Wf : (ncol < 8) ? Wi : (ncol < 12) ? Wu : Wo;
    const int h  = ncol & 3;                    // column within gate (W_g is [68,4] row-major)
    const int kb = (lane & 16) ? 16 : 0;
    v16h b0, b1;
#pragma unroll
    for (int k = 0; k < 16; ++k) {
        b0[k] = (_Float16)Wp[(kb + k) * 4 + h];        // K chunk 0..31
        b1[k] = (_Float16)Wp[(kb + k + 32) * 4 + h];   // K chunk 32..63
    }

    // ---- A-fragment lane mapping (16x32 f16):
    // lanes 0-15: row=lane, K in {0..15}; lanes 16-31: row=lane-16, K in {16..31}
    const int rsub = lane & 15;
    const int kA   = (lane & 16) ? 16 : 0;
    const int msub = (lane & 16) ? 8 : 0;

    for (int tile = wave; tile < Mtiles; tile += nwaves) {
        const int m0 = tile * 16;
        const float* xr = x + (size_t)(m0 + rsub) * 64 + kA;
        v16h a0, a1;
#pragma unroll
        for (int k = 0; k < 8; ++k) {
            a0[k]     = (_Float16)xr[k];        // K = kA + k
            a0[k + 8] = (_Float16)xr[8 + k];    // K = kA + 8 + k
            a1[k]     = (_Float16)xr[32 + k];   // K = 32 + kA + k
            a1[k + 8] = (_Float16)xr[40 + k];   // K = 32 + kA + 8 + k
        }
        v8f c = {};
        c = __builtin_amdgcn_wmma_f32_16x16x32_f16(false, a0, false, b0, (short)0, c, false, false);
        c = __builtin_amdgcn_wmma_f32_16x16x32_f16(false, a1, false, b1, (short)0, c, false, false);

        // D layout: VGPR r -> row (r + msub), col = ncol
        float* pr = pre + (size_t)(m0 + msub) * 16 + ncol;
#pragma unroll
        for (int r = 0; r < 8; ++r)
            pr[r * 16] = c[r];
    }
}

// ---------------------------------------------------------------------------
// Phase 2: sequential scan over T. 4 lanes per batch element (one per gate).
// Statevector of 4 qubits = 16 complex amps, fully in registers, unrolled.
// State index: i = q0*8 + q1*4 + q2*2 + q3  (wire w -> mask 8>>w)
// ---------------------------------------------------------------------------
__device__ __forceinline__ void apply_rx(float re[16], float im[16],
                                         float c, float s, int mask)
{
#pragma unroll
    for (int i = 0; i < 16; ++i) {
        if (i & mask) continue;
        const int j = i | mask;
        const float a0r = re[i], a0i = im[i], a1r = re[j], a1i = im[j];
        re[i] = c * a0r + s * a1i;   // c*a0 - i*s*a1
        im[i] = c * a0i - s * a1r;
        re[j] = c * a1r + s * a0i;   // -i*s*a0 + c*a1
        im[j] = c * a1i - s * a0r;
    }
}

__device__ __forceinline__ void cnot_(float re[16], float im[16], int cmask, int tmask)
{
#pragma unroll
    for (int i = 0; i < 16; ++i) {
        if ((i & cmask) && !(i & tmask)) {
            const int j = i | tmask;
            float tr = re[i], ti = im[i];
            re[i] = re[j]; im[i] = im[j];
            re[j] = tr;    im[j] = ti;
        }
    }
}

__device__ __forceinline__ float sigm_(float x) { return 1.0f / (1.0f + __expf(-x)); }

__global__ __launch_bounds__(256)
void qlstm_scan(const float* __restrict__ pre,
                const float* __restrict__ Wf, const float* __restrict__ Wi,
                const float* __restrict__ Wu, const float* __restrict__ Wo,
                const float* __restrict__ bf, const float* __restrict__ bi,
                const float* __restrict__ bu, const float* __restrict__ bo,
                const float* __restrict__ thf, const float* __restrict__ thi,
                const float* __restrict__ thu, const float* __restrict__ tho,
                const float* __restrict__ Wtag, const float* __restrict__ btag,
                float* __restrict__ out, int T, int Bn)
{
    const int tid  = blockIdx.x * blockDim.x + threadIdx.x;
    const int g    = tid & 3;      // gate: 0=f 1=i 2=u 3=o
    const int b    = tid >> 2;     // batch element
    if (b >= Bn) return;
    const int lane  = threadIdx.x & 31;
    const int qbase = lane & ~3;

    const float* Wsel  = (g == 0) ? Wf  : (g == 1) ? Wi  : (g == 2) ? Wu  : Wo;
    const float* bsel  = (g == 0) ? bf  : (g == 1) ? bi  : (g == 2) ? bu  : bo;
    const float* thsel = (g == 0) ? thf : (g == 1) ? thi : (g == 2) ? thu : tho;

    // Recurrent weights W_g[64:68, 0:4], bias, precomputed cos/sin(theta/2)
    float Wh[4][4];
#pragma unroll
    for (int j = 0; j < 4; ++j)
#pragma unroll
        for (int hh = 0; hh < 4; ++hh)
            Wh[j][hh] = Wsel[(64 + j) * 4 + hh];
    float bias[4], thc[4], ths[4];
#pragma unroll
    for (int hh = 0; hh < 4; ++hh) {
        bias[hh] = bsel[hh];
        const float a = thsel[hh] * 0.5f;
        thc[hh] = __cosf(a);
        ths[hh] = __sinf(a);
    }
    // Tag head (uniform addresses -> scalar loads)
    float Wt[40], bt[10];
#pragma unroll
    for (int k = 0; k < 40; ++k) Wt[k] = Wtag[k];
#pragma unroll
    for (int k = 0; k < 10; ++k) bt[k] = btag[k];

    float hx[4] = {0.f, 0.f, 0.f, 0.f};
    float cx[4] = {0.f, 0.f, 0.f, 0.f};

    for (int t = 0; t < T; ++t) {
        const size_t r = (size_t)t * Bn + b;
        const float* pr = pre + r * 16 + g * 4;
        if (t + 1 < T)
            __builtin_prefetch(pre + (r + Bn) * 16 + g * 4, 0, 0);  // global_prefetch_b8

        // angles = pre + bias + hx @ Wh
        float ang[4];
#pragma unroll
        for (int hh = 0; hh < 4; ++hh)
            ang[hh] = pr[hh] + bias[hh]
                    + hx[0] * Wh[0][hh] + hx[1] * Wh[1][hh]
                    + hx[2] * Wh[2][hh] + hx[3] * Wh[3][hh];

        // 4-qubit statevector sim
        float re[16], im[16];
#pragma unroll
        for (int i = 0; i < 16; ++i) { re[i] = 0.f; im[i] = 0.f; }
        re[0] = 1.f;
#pragma unroll
        for (int w = 0; w < 4; ++w) {
            const float a = ang[w] * 0.5f;
            apply_rx(re, im, __cosf(a), __sinf(a), 8 >> w);
        }
#pragma unroll
        for (int w = 0; w < 4; ++w)
            apply_rx(re, im, thc[w], ths[w], 8 >> w);
        cnot_(re, im, 8, 4);   // (0,1)
        cnot_(re, im, 4, 2);   // (1,2)
        cnot_(re, im, 2, 1);   // (2,3)
        cnot_(re, im, 1, 8);   // (3,0)

        float q[4] = {0.f, 0.f, 0.f, 0.f};
#pragma unroll
        for (int i = 0; i < 16; ++i) {
            const float p = re[i] * re[i] + im[i] * im[i];
            q[0] += (i & 8) ? -p : p;
            q[1] += (i & 4) ? -p : p;
            q[2] += (i & 2) ? -p : p;
            q[3] += (i & 1) ? -p : p;
        }

        // gate nonlinearity (u-gate uses tanh)
        float act[4];
#pragma unroll
        for (int hh = 0; hh < 4; ++hh)
            act[hh] = (g == 2) ? tanhf(q[hh]) : sigm_(q[hh]);

        // quad exchange: every lane needs all four gate vectors
        float fv[4], iv[4], gv[4], ov[4];
#pragma unroll
        for (int hh = 0; hh < 4; ++hh) {
            fv[hh] = __shfl(act[hh], qbase + 0, 32);
            iv[hh] = __shfl(act[hh], qbase + 1, 32);
            gv[hh] = __shfl(act[hh], qbase + 2, 32);
            ov[hh] = __shfl(act[hh], qbase + 3, 32);
        }
#pragma unroll
        for (int hh = 0; hh < 4; ++hh) {
            cx[hh] = fv[hh] * cx[hh] + iv[hh] * gv[hh];
            hx[hh] = ov[hh] * tanhf(cx[hh]);
        }

        // tag projection + log_softmax (one lane per quad writes)
        if (g == 0) {
            float l[10];
#pragma unroll
            for (int k = 0; k < 10; ++k)
                l[k] = bt[k] + hx[0] * Wt[k]      + hx[1] * Wt[10 + k]
                             + hx[2] * Wt[20 + k] + hx[3] * Wt[30 + k];
            float m = l[0];
#pragma unroll
            for (int k = 1; k < 10; ++k) m = fmaxf(m, l[k]);
            float ssum = 0.f;
#pragma unroll
            for (int k = 0; k < 10; ++k) ssum += __expf(l[k] - m);
            const float lse = m + __logf(ssum);
            float* op = out + r * 10;
#pragma unroll
            for (int k = 0; k < 10; ++k) op[k] = l[k] - lse;
        }
    }
}

// ---------------------------------------------------------------------------
extern "C" void kernel_launch(void* const* d_in, const int* in_sizes, int n_in,
                              void* d_out, int out_size, void* d_ws, size_t ws_size,
                              hipStream_t stream)
{
    const float* x    = (const float*)d_in[0];
    const float* Wf   = (const float*)d_in[1];
    const float* bf   = (const float*)d_in[2];
    const float* Wi_  = (const float*)d_in[3];
    const float* bi   = (const float*)d_in[4];
    const float* Wu   = (const float*)d_in[5];
    const float* bu   = (const float*)d_in[6];
    const float* Wo   = (const float*)d_in[7];
    const float* bo   = (const float*)d_in[8];
    const float* thf  = (const float*)d_in[9];
    const float* thi  = (const float*)d_in[10];
    const float* thu  = (const float*)d_in[11];
    const float* tho  = (const float*)d_in[12];
    const float* Wtag = (const float*)d_in[13];
    const float* btag = (const float*)d_in[14];
    float* out = (float*)d_out;

    const int D  = 64;
    const int TB = in_sizes[0] / D;   // T*B flattened rows
    const int Bn = 2048;              // batch from setup_inputs
    const int T  = TB / Bn;

    float* pre = (float*)d_ws;        // TB * 16 floats (32 MiB)

    // Phase 1: big parallel WMMA GEMM (HBM-bound, ~9 us of traffic at 23.3 TB/s)
    const int Mtiles = TB / 16;       // 32768 tiles of 16 rows
    qlstm_pre_gemm<<<512, 256, 0, stream>>>(x, Wf, Wi_, Wu, Wo, pre, Mtiles);

    // Phase 2: latency-bound recurrent scan, 4 lanes per batch element
    const int threads2 = Bn * 4;      // 8192
    qlstm_scan<<<(threads2 + 255) / 256, 256, 0, stream>>>(
        pre, Wf, Wi_, Wu, Wo, bf, bi, bu, bo,
        thf, thi, thu, tho, Wtag, btag, out, T, Bn);
}
